// OneToManyEncoder_44341242364564
// MI455X (gfx1250) — matrix operand order repaired and checked by reading, not compiled, not run on hardware
//
#include <hip/hip_runtime.h>

// ---------------------------------------------------------------------------
// OneToManyEncoder for MI455X (gfx1250), wave32 + WMMA f16 + TDM path.
//   B_A=16384, B_B=100000, D=256, M=50, d_in=512, d_hidden=512, d_out=256
// Pipeline:
//   1) convert W1/W2 f32->f16 into workspace
//   2) gather + masked-mean + pack combined=[a|agg] as f16 [16384 x 512]
//   3) h = relu(combined @ W1 + b1)  (f16 WMMA, f32 accum, f16 out)
//   4) out = h @ W2 + b2             (f16 WMMA, f32 accum, f32 out)
// A-tiles are staged into LDS by the Tensor Data Mover (tensor_load_to_lds,
// TENSORcnt); B-tiles are staged transposed via VGPRs (TDM can't transpose).
// ---------------------------------------------------------------------------

typedef __attribute__((ext_vector_type(16))) _Float16 v16h;
typedef __attribute__((ext_vector_type(8)))  _Float16 v8h;
typedef __attribute__((ext_vector_type(4)))  _Float16 v4h;
typedef __attribute__((ext_vector_type(8)))  float    v8f;
typedef __attribute__((ext_vector_type(4)))  unsigned int v4u;
typedef __attribute__((ext_vector_type(8)))  int      v8i;
typedef __attribute__((ext_vector_type(4)))  int      v4i;

union F16x16 { v16h v; v8h h[2]; };

#define BA_ROWS 16384
#define BB_ROWS 100000
#define DMODEL  256
#define MMATCH  50
#define DIN     512   // 2*D
#define DHID    512   // 2*D
#define DOUT    256   // D

#if defined(__has_builtin)
#if __has_builtin(__builtin_amdgcn_tensor_load_to_lds) && \
    __has_builtin(__builtin_amdgcn_s_wait_tensorcnt)
#define USE_TDM 1
#endif
#endif

// ---------------------------------------------------------------------------
// f32 -> f16 elementwise convert (n must be a multiple of 1024)
// ---------------------------------------------------------------------------
__global__ __launch_bounds__(256) void cvt_f32_f16_kernel(
    const float* __restrict__ src, _Float16* __restrict__ dst, int n) {
  int i = (blockIdx.x * 256 + threadIdx.x) * 4;
  if (i >= n) return;
  float4 x = *(const float4*)(src + i);
  v4h o;
  o[0] = (_Float16)x.x; o[1] = (_Float16)x.y;
  o[2] = (_Float16)x.z; o[3] = (_Float16)x.w;
  *(v4h*)(dst + i) = o;
}

// ---------------------------------------------------------------------------
// Gather + masked mean + pack: one wave (32 lanes) per row of table A.
// Each lane owns 8 contiguous floats (2 x float4) of the 256-wide row.
// table_b (102 MB) is L2-resident on MI455X (192 MB L2), so the random
// gather is served from L2 after first touch.
// ---------------------------------------------------------------------------
__global__ __launch_bounds__(256) void gather_pack_kernel(
    const float* __restrict__ ta, const float* __restrict__ tb,
    const int* __restrict__ mi, _Float16* __restrict__ combined) {
  const int wave = threadIdx.x >> 5;
  const int lane = threadIdx.x & 31;
  const int row  = blockIdx.x * 8 + wave;
  const int col  = lane * 8;

  float acc[8];
#pragma unroll
  for (int e = 0; e < 8; ++e) acc[e] = 0.0f;
  int cnt = 0;

  const int* mrow = mi + row * MMATCH;
#pragma unroll 2
  for (int m = 0; m < MMATCH; ++m) {
    int j = mrow[m];                       // same for all lanes in the wave
    if (j >= 0) {
      ++cnt;
      const float4* p = (const float4*)(tb + (size_t)j * DMODEL + col);
      float4 x0 = p[0], x1 = p[1];
      acc[0] += x0.x; acc[1] += x0.y; acc[2] += x0.z; acc[3] += x0.w;
      acc[4] += x1.x; acc[5] += x1.y; acc[6] += x1.z; acc[7] += x1.w;
    }
  }
  const float inv = 1.0f / fmaxf((float)cnt, 1.0f);

  const float4* pa = (const float4*)(ta + (size_t)row * DMODEL + col);
  float4 a0 = pa[0], a1 = pa[1];

  v8h ha, hg;
  ha[0] = (_Float16)a0.x; ha[1] = (_Float16)a0.y;
  ha[2] = (_Float16)a0.z; ha[3] = (_Float16)a0.w;
  ha[4] = (_Float16)a1.x; ha[5] = (_Float16)a1.y;
  ha[6] = (_Float16)a1.z; ha[7] = (_Float16)a1.w;
#pragma unroll
  for (int e = 0; e < 8; ++e) hg[e] = (_Float16)(acc[e] * inv);

  _Float16* crow = combined + (size_t)row * DIN;
  *(v8h*)(crow + col)          = ha;
  *(v8h*)(crow + DMODEL + col) = hg;
}

// ---------------------------------------------------------------------------
// LDS-tiled WMMA GEMM:  C[16384 x NDIM] = act(A[16384 x K] * B[K x NDIM] + bias)
//   Workgroup: 256 threads = 8 waves; tile BM=128, BN=128, BK=32.
//   Wave w computes a 64x32 output tile: 4 (row) x 2 (col) 16x16 WMMA accs.
//   A tile: staged by TDM (tensor_load_to_lds, wave 0 issues + waits).
//   B tile: staged transposed (sB[n][k]) so a B fragment is one contiguous
//   32-byte LDS read per lane in contiguous k order (k = 16*hi + s), as A uses.
// ---------------------------------------------------------------------------
template <int NDIM, bool RELU, bool OUT_F16>
__global__ __launch_bounds__(256) void wmma_gemm_kernel(
    const _Float16* __restrict__ A, const _Float16* __restrict__ Bm,
    const float* __restrict__ bias, void* __restrict__ Cout, int K) {
  __shared__ __align__(64) _Float16 sA[128 * 32];  // [row][k]
  __shared__ __align__(64) _Float16 sB[128 * 32];  // [n][k]  (transposed)

  const int t       = threadIdx.x;
  const int wave    = t >> 5;
  const int lane    = t & 31;
  const int l15     = lane & 15;
  const int hi      = lane >> 4;          // 0: lanes 0-15, 1: lanes 16-31
  const int rowBase = blockIdx.x * 128;
  const int colBase = blockIdx.y * 128;
  const int rw      = (wave >> 2) * 64;   // wave row offset in tile
  const int cw      = (wave & 3) * 32;    // wave col offset in tile
  const int kbA     = hi * 16;            // A-frag K sub-offset (contiguous k = 16*hi + s)
  const int kbB     = hi * 16;            // B-frag K sub-offset (contiguous k = 16*hi + s)

  v8f acc[4][2];
#pragma unroll
  for (int i = 0; i < 4; ++i)
#pragma unroll
    for (int j = 0; j < 2; ++j) acc[i][j] = (v8f)(0.0f);

  const int KT = K / 32;
  for (int kt = 0; kt < KT; ++kt) {
    const int k0 = kt * 32;
    __syncthreads();  // previous iteration's LDS reads complete

#ifdef USE_TDM
    // --- A tile via Tensor Data Mover: 128x32 f16 2D tile ----------------
    if (wave == 0) {
      const unsigned long long ga =
          (unsigned long long)(size_t)(A + (size_t)rowBase * K + k0);
      v4u g0;
      g0[0] = 1u;                                        // count=1 (valid D#)
      g0[1] = (unsigned)(size_t)&sA[0];                  // lds_addr (bytes)
      g0[2] = (unsigned)ga;                              // global_addr[31:0]
      g0[3] = (unsigned)((ga >> 32) & 0x1FFFFFFull)      // global_addr[56:32]
              | (2u << 30);                              // type=2 ("image")
      v8i g1;
      g1[0] = 1 << 16;                  // workgroup_mask=0, data_size=1 (2B)
      g1[1] = (K & 0xFFFF) << 16;       // tensor_dim0[15:0] = K
      g1[2] = ((K >> 16) & 0xFFFF)      // tensor_dim0[31:16]
              | (BA_ROWS << 16);        // tensor_dim1[15:0] = 16384
      g1[3] = (BA_ROWS >> 16)           // tensor_dim1[31:16]
              | (32 << 16);             // tile_dim0 = 32 (K per tile)
      g1[4] = 128;                      // tile_dim1 = 128 rows, tile_dim2=0
      g1[5] = K;                        // tensor_dim0_stride[31:0] = K
      g1[6] = 0;                        // stride[47:32]=0, dim1_stride lo=0
      g1[7] = 0;
      v4i z4 = {};
      v8i z8 = {};
      // 6-arg form (clang-23 / therock headers): groups 2,3 unused (2D),
      // trailing int32x8 zero-filled, cpol=0.
      __builtin_amdgcn_tensor_load_to_lds(g0, g1, z4, z4, z8, 0);
    }
#else
    // --- fallback: A tile via VGPRs (16B per thread x2) ------------------
#pragma unroll
    for (int i = 0; i < 2; ++i) {
      int c  = t + i * 256;
      int ar = c >> 2;
      int ko = (c & 3) * 8;
      *(v8h*)&sA[ar * 32 + ko] =
          *(const v8h*)(A + (size_t)(rowBase + ar) * K + k0 + ko);
    }
#endif

    // --- B tile transposed: read two adjacent k-rows coalesced, ---------
    //     pack (k,k+1) half-pairs, scatter as 8 x b32 per thread ---------
    {
      int k2 = (t >> 4) * 2;            // even k in [0,32)
      int no = (t & 15) * 8;            // n offset in [0,128)
      const _Float16* bp = Bm + (size_t)(k0 + k2) * NDIM + colBase + no;
      v8h r0 = *(const v8h*)bp;
      v8h r1 = *(const v8h*)(bp + NDIM);
#pragma unroll
      for (int e = 0; e < 8; ++e) {
        union { _Float16 h[2]; unsigned u; } pk;
        pk.h[0] = r0[e];
        pk.h[1] = r1[e];
        *(unsigned*)&sB[(no + e) * 32 + k2] = pk.u;  // 4B aligned (k2 even)
      }
    }
    // prefetch next weight K-chunk into cache (global_prefetch_b8)
    if (kt + 1 < KT) {
      __builtin_prefetch(
          Bm + (size_t)(k0 + 32 + (t & 31)) * NDIM + colBase + (t >> 5) * 16,
          0, 0);
    }
#ifdef USE_TDM
    if (wave == 0) __builtin_amdgcn_s_wait_tensorcnt(0);  // A tile landed
#endif
    __syncthreads();

    // --- fragments, contiguous k order (k = 16*hi + s) in A and B ---------------------------------
    F16x16 af[4];
#pragma unroll
    for (int i = 0; i < 4; ++i) {
      int ar = rw + i * 16 + l15;
      af[i].h[0] = *(const v8h*)&sA[ar * 32 + kbA];        // K 0-7 / 16-23
      af[i].h[1] = *(const v8h*)&sA[ar * 32 + kbA + 8];    // K 8-15 / 24-31
    }
    v16h bf[2];
#pragma unroll
    for (int j = 0; j < 2; ++j) {
      int bc = cw + j * 16 + l15;
      bf[j] = *(const v16h*)&sB[bc * 32 + kbB];            // K 0-15 / 16-31
    }
    // --- 8 WMMAs: D = A x B + C -----------------------------------------
#pragma unroll
    for (int i = 0; i < 4; ++i)
#pragma unroll
      for (int j = 0; j < 2; ++j)
        acc[i][j] = __builtin_amdgcn_wmma_f32_16x16x32_f16(
            false, af[i].v, false, bf[j], (short)0, acc[i][j], false, false);
  }

  // --- epilogue: bias (+ReLU), scatter per f32 C/D lane layout ----------
#pragma unroll
  for (int j = 0; j < 2; ++j) {
    const int col_g = colBase + cw + j * 16 + l15;
    const float bv  = bias[col_g];
#pragma unroll
    for (int i = 0; i < 4; ++i) {
#pragma unroll
      for (int v = 0; v < 8; ++v) {
        int row_g = rowBase + rw + i * 16 + v + hi * 8;
        float x = acc[i][j][v] + bv;
        if (RELU) x = fmaxf(x, 0.0f);
        if constexpr (OUT_F16) {
          ((_Float16*)Cout)[(size_t)row_g * NDIM + col_g] = (_Float16)x;
        } else {
          ((float*)Cout)[(size_t)row_g * NDIM + col_g] = x;
        }
      }
    }
  }
}

// ---------------------------------------------------------------------------
// Launch
// ---------------------------------------------------------------------------
extern "C" void kernel_launch(void* const* d_in, const int* in_sizes, int n_in,
                              void* d_out, int out_size, void* d_ws,
                              size_t ws_size, hipStream_t stream) {
  const float* ta = (const float*)d_in[0];   // [16384,256]
  const float* tb = (const float*)d_in[1];   // [100000,256]
  const int*   mi = (const int*)d_in[2];     // [16384,50]
  const float* W1 = (const float*)d_in[3];   // [512,512]
  const float* b1 = (const float*)d_in[4];   // [512]
  const float* W2 = (const float*)d_in[5];   // [512,256]
  const float* b2 = (const float*)d_in[6];   // [256]

  // Workspace layout (f16): combined 16MB | h 16MB | W1h 512KB | W2h 256KB
  _Float16* combined = (_Float16*)d_ws;
  _Float16* hbuf     = combined + (size_t)BA_ROWS * DIN;
  _Float16* W1h      = hbuf + (size_t)BA_ROWS * DHID;
  _Float16* W2h      = W1h + (size_t)DIN * DHID;

  // 1) weights to f16
  cvt_f32_f16_kernel<<<(DIN * DHID) / 1024, 256, 0, stream>>>(W1, W1h,
                                                              DIN * DHID);
  cvt_f32_f16_kernel<<<(DHID * DOUT) / 1024, 256, 0, stream>>>(W2, W2h,
                                                               DHID * DOUT);

  // 2) gather + masked mean + pack combined
  gather_pack_kernel<<<BA_ROWS / 8, 256, 0, stream>>>(ta, tb, mi, combined);

  // 3) h = relu(combined @ W1 + b1)   [16384 x 512] f16
  {
    dim3 grid(BA_ROWS / 128, DHID / 128);
    wmma_gemm_kernel<DHID, true, true>
        <<<grid, 256, 0, stream>>>(combined, W1h, b1, hbuf, DIN);
  }

  // 4) out = h @ W2 + b2              [16384 x 256] f32
  {
    dim3 grid(BA_ROWS / 128, DOUT / 128);
    wmma_gemm_kernel<DOUT, false, false>
        <<<grid, 256, 0, stream>>>(hbuf, W2h, b2, d_out, DHID);
  }
}
